// OnlineTripleLoss_48601849921735
// MI455X (gfx1250) — matrix operand order repaired and checked by reading, not compiled, not run on hardware
//
#include <hip/hip_runtime.h>
#include <hip/hip_bf16.h>

typedef __attribute__((ext_vector_type(16))) _Float16 v16h;
typedef __attribute__((ext_vector_type(8)))  _Float16 v8h;
typedef __attribute__((ext_vector_type(8)))  float    v8f;

#define TL_N     8192
#define TL_D     256
#define TL_C     512
#define TILE     128
#define PAD      264          // 256 + 8 f16 pad to stagger LDS banks
#define NTILES   64           // 8192 / 128
#define MARGIN_F 0.2f

// ---------- workspace layout (bytes) ----------
// xnh     : 0        , 8192*256*2 = 4194304
// dmink   : 4194304  , 8192*4     = 32768
// clscnt  : 4227072  , 512*4      = 2048
// partial : 4229120  , 4096*4     = 16384
#define WS_XNH    0
#define WS_DMINK  4194304
#define WS_CLSCNT 4227072
#define WS_PART   4229120

__device__ __forceinline__ unsigned encodeKey(float f) {
    unsigned u = __float_as_uint(f);
    return (u & 0x80000000u) ? ~u : (u | 0x80000000u);
}
__device__ __forceinline__ float decodeKey(unsigned k) {
    return (k & 0x80000000u) ? __uint_as_float(k & 0x7FFFFFFFu)
                             : __uint_as_float(~k);
}

// ---- CDNA5 async direct-to-LDS staging (ASYNCcnt path, no VGPR round trip) ----
__device__ __forceinline__ void async_ld16(unsigned lds_off, const void* gptr) {
    asm volatile("global_load_async_to_lds_b128 %0, %1, off"
                 :: "v"(lds_off), "v"(gptr) : "memory");
}
__device__ __forceinline__ void wait_async0() {
    asm volatile("s_wait_asynccnt 0" ::: "memory");
}

// ---------------- init ----------------
__global__ void tl_init(unsigned* __restrict__ dmink, int* __restrict__ clscnt,
                        float* __restrict__ partial) {
    int i = blockIdx.x * 256 + threadIdx.x;
    if (i < TL_N)  dmink[i]   = 0xFFFFFFFFu;   // +inf key
    if (i < TL_C)  clscnt[i]  = 0;
    if (i < 4096)  partial[i] = 0.0f;
}

// ---------------- class histogram ----------------
__global__ void tl_count(const int* __restrict__ labels, int* __restrict__ clscnt) {
    int i = blockIdx.x * 256 + threadIdx.x;
    if (i < TL_N) atomicAdd(&clscnt[labels[i]], 1);
}

// ---------------- normalize rows -> f16 ----------------
__global__ __launch_bounds__(256) void tl_norm(const float* __restrict__ emb,
                                               _Float16* __restrict__ xnh) {
    int row  = blockIdx.x * 8 + (threadIdx.x >> 5);
    int lane = threadIdx.x & 31;
    const float4* p = (const float4*)(emb + (size_t)row * TL_D) + lane * 2;
    float4 a = p[0], b = p[1];
    float s = a.x*a.x + a.y*a.y + a.z*a.z + a.w*a.w
            + b.x*b.x + b.y*b.y + b.z*b.z + b.w*b.w;
    for (int m = 1; m < 32; m <<= 1) s += __shfl_xor(s, m, 32);
    float scale = 1.0f / fmaxf(sqrtf(s), 1e-8f);
    v8h o;
    o[0] = (_Float16)(a.x * scale); o[1] = (_Float16)(a.y * scale);
    o[2] = (_Float16)(a.z * scale); o[3] = (_Float16)(a.w * scale);
    o[4] = (_Float16)(b.x * scale); o[5] = (_Float16)(b.y * scale);
    o[6] = (_Float16)(b.z * scale); o[7] = (_Float16)(b.w * scale);
    *(v8h*)(xnh + (size_t)row * TL_D + lane * 8) = o;
}

// ---------------- shared GEMM-core helpers ----------------
__device__ __forceinline__ void tl_load_tiles(const _Float16* __restrict__ xnh,
                                              int rowBase, int colBase,
                                              _Float16 (*lA)[PAD], _Float16 (*lB)[PAD],
                                              const int* __restrict__ labels,
                                              int* sLabA, int* sLabB) {
    int t = threadIdx.x;
#pragma unroll
    for (int s = 0; s < 16; ++s) {
        int idx = s * 256 + t;            // 0..4095 16-byte chunks per tile
        int r   = idx >> 5;               // 32 chunks per 256-elem row
        int c   = (idx & 31) * 8;
        async_ld16((unsigned)(size_t)&lA[r][c],
                   xnh + (size_t)(rowBase + r) * TL_D + c);
        async_ld16((unsigned)(size_t)&lB[r][c],
                   xnh + (size_t)(colBase + r) * TL_D + c);
    }
    if (t < TILE) { sLabA[t] = labels[rowBase + t]; sLabB[t] = labels[colBase + t]; }
    wait_async0();                        // this wave's async fills complete
}

// 16x128 strip per wave: 8 accumulators of 16x16, K=256 in 8 steps of 32.
__device__ __forceinline__ void tl_mma(const _Float16 (*lA)[PAD], const _Float16 (*lB)[PAD],
                                       v8f acc[8], int m0, int h16, int lm) {
#pragma unroll
    for (int ks = 0; ks < 8; ++ks) {
        const int k0 = ks * 32;
        // A frag (16-bit A layout): lane holds row m0+lm; K = k0+8*h16..+7 then k0+16+8*h16..+7
        v8h alo = *(const v8h*)&lA[m0 + lm][k0 + h16 * 8];
        v8h ahi = *(const v8h*)&lA[m0 + lm][k0 + 16 + h16 * 8];
        v16h af = __builtin_shufflevector(alo, ahi, 0,1,2,3,4,5,6,7,8,9,10,11,12,13,14,15);
#pragma unroll
        for (int nt = 0; nt < 8; ++nt) {
            // B frag: lane holds column nt*16+lm; K = k0+8*h16..+7 then k0+16+8*h16..+7
            v8h blo = *(const v8h*)&lB[nt * 16 + lm][k0 + h16 * 8];
            v8h bhi = *(const v8h*)&lB[nt * 16 + lm][k0 + 16 + h16 * 8];
            v16h bf = __builtin_shufflevector(blo, bhi, 0,1,2,3,4,5,6,7,8,9,10,11,12,13,14,15);
            acc[nt] = __builtin_amdgcn_wmma_f32_16x16x32_f16(
                false, af, false, bf, (short)0, acc[nt], false, false);
        }
    }
}

// ---------------- pass 1: per-row min over negatives ----------------
__global__ __launch_bounds__(256) void tl_minneg(const _Float16* __restrict__ xnh,
                                                 const int* __restrict__ labels,
                                                 unsigned* __restrict__ dmink) {
    __shared__ _Float16 lA[TILE][PAD];
    __shared__ _Float16 lB[TILE][PAD];
    __shared__ int sLabA[TILE], sLabB[TILE];

    const int rowBase = blockIdx.y * TILE;
    const int colBase = blockIdx.x * TILE;
    tl_load_tiles(xnh, rowBase, colBase, lA, lB, labels, sLabA, sLabB);
    __syncthreads();

    const int wave = threadIdx.x >> 5, lane = threadIdx.x & 31;
    const int h16 = lane >> 4, lm = lane & 15;
    const int m0 = wave * 16;

    v8f acc[8];
    const v8f zero = {0.f,0.f,0.f,0.f,0.f,0.f,0.f,0.f};
#pragma unroll
    for (int nt = 0; nt < 8; ++nt) acc[nt] = zero;

    tl_mma(lA, lB, acc, m0, h16, lm);

#pragma unroll
    for (int r = 0; r < 8; ++r) {
        const int iloc = m0 + r + 8 * h16;   // C/D layout: VGPR r -> M=r (lanes 0-15), M=8+r (16-31)
        const int li = sLabA[iloc];
        float mn = 3.0e38f;
#pragma unroll
        for (int nt = 0; nt < 8; ++nt) {
            const int jloc = nt * 16 + lm;
            if (sLabB[jloc] != li) {
                float d = 1.0f - acc[nt][r];
                mn = fminf(mn, d);
            }
        }
        for (int m = 1; m < 16; m <<= 1) mn = fminf(mn, __shfl_xor(mn, m, 32));  // within half
        if (lm == 0 && mn < 1.0e38f)
            atomicMin(&dmink[rowBase + iloc], encodeKey(mn));   // order-independent => deterministic
    }
}

// ---------------- pass 2: masked hinge-loss sum ----------------
__global__ __launch_bounds__(256) void tl_loss(const _Float16* __restrict__ xnh,
                                               const int* __restrict__ labels,
                                               const unsigned* __restrict__ dmink,
                                               const int* __restrict__ clscnt,
                                               float* __restrict__ partial) {
    const int ct = blockIdx.x, rt = blockIdx.y;
    if (ct < rt) return;                    // strictly-lower tiles have no a<p pairs

    __shared__ _Float16 lA[TILE][PAD];
    __shared__ _Float16 lB[TILE][PAD];
    __shared__ int sLabA[TILE], sLabB[TILE];
    __shared__ float sred[8];

    const int rowBase = rt * TILE;
    const int colBase = ct * TILE;
    tl_load_tiles(xnh, rowBase, colBase, lA, lB, labels, sLabA, sLabB);
    __syncthreads();

    const int wave = threadIdx.x >> 5, lane = threadIdx.x & 31;
    const int h16 = lane >> 4, lm = lane & 15;
    const int m0 = wave * 16;

    v8f acc[8];
    const v8f zero = {0.f,0.f,0.f,0.f,0.f,0.f,0.f,0.f};
#pragma unroll
    for (int nt = 0; nt < 8; ++nt) acc[nt] = zero;

    tl_mma(lA, lB, acc, m0, h16, lm);

    float psum = 0.0f;
#pragma unroll
    for (int r = 0; r < 8; ++r) {
        const int iloc = m0 + r + 8 * h16;
        const int iG = rowBase + iloc;
        const int li = sLabA[iloc];
        if (clscnt[li] < TL_N) {                      // has_neg[a]
            const float dmin = decodeKey(dmink[iG]);
#pragma unroll
            for (int nt = 0; nt < 8; ++nt) {
                const int jloc = nt * 16 + lm;
                const int jG = colBase + jloc;
                if (jG > iG && sLabB[jloc] == li)
                    psum += fmaxf(1.0f - acc[nt][r] - dmin + MARGIN_F, 0.0f);
            }
        }
    }
    for (int m = 1; m < 32; m <<= 1) psum += __shfl_xor(psum, m, 32);
    if (lane == 0) sred[wave] = psum;
    __syncthreads();
    if (threadIdx.x == 0) {
        float t = 0.f;
        for (int w = 0; w < 8; ++w) t += sred[w];
        partial[rt * NTILES + ct] = t;               // deterministic (no float atomics)
    }
}

// ---------------- finalize: count + loss ----------------
__global__ __launch_bounds__(256) void tl_final(const int* __restrict__ clscnt,
                                                const float* __restrict__ partial,
                                                float* __restrict__ out) {
    __shared__ float ssum[256];
    __shared__ unsigned long long scnt[256];
    const int t = threadIdx.x;
    float s = 0.0f;
    for (int i = t; i < 4096; i += 256) s += partial[i];
    unsigned long long c = 0;
    for (int i = t; i < TL_C; i += 256) {
        long long n = clscnt[i];
        if (n < TL_N) c += (unsigned long long)(n * (n - 1) / 2);
    }
    ssum[t] = s; scnt[t] = c;
    __syncthreads();
    for (int off = 128; off > 0; off >>= 1) {
        if (t < off) { ssum[t] += ssum[t + off]; scnt[t] += scnt[t + off]; }
        __syncthreads();
    }
    if (t == 0) {
        float cnt = (float)scnt[0];
        out[0] = (scnt[0] > 0) ? (ssum[0] / cnt) : 0.0f;
        out[1] = cnt;
    }
}

extern "C" void kernel_launch(void* const* d_in, const int* in_sizes, int n_in,
                              void* d_out, int out_size, void* d_ws, size_t ws_size,
                              hipStream_t stream) {
    const float* emb    = (const float*)d_in[0];
    const int*   labels = (const int*)d_in[1];
    char* ws = (char*)d_ws;
    _Float16* xnh     = (_Float16*)(ws + WS_XNH);
    unsigned* dmink   = (unsigned*)(ws + WS_DMINK);
    int*      clscnt  = (int*)(ws + WS_CLSCNT);
    float*    partial = (float*)(ws + WS_PART);
    float*    out     = (float*)d_out;

    tl_init <<<32, 256, 0, stream>>>(dmink, clscnt, partial);
    tl_count<<<32, 256, 0, stream>>>(labels, clscnt);
    tl_norm <<<TL_N / 8, 256, 0, stream>>>(emb, xnh);

    dim3 grid(NTILES, NTILES);
    tl_minneg<<<grid, 256, 0, stream>>>(xnh, labels, dmink);
    tl_loss  <<<grid, 256, 0, stream>>>(xnh, labels, dmink, clscnt, partial);
    tl_final <<<1, 256, 0, stream>>>(clscnt, partial, out);
}
